// NeuralImageFunction_53747220742611
// MI455X (gfx1250) — compile-verified, matrix operand order checked by reading, not run on hardware
//
#include <hip/hip_runtime.h>
#include <math.h>

typedef _Float16 h8   __attribute__((ext_vector_type(8)));
typedef _Float16 v16h __attribute__((ext_vector_type(16)));
typedef float    v8f  __attribute__((ext_vector_type(8)));

#define WMMA_F16(A,B,C) \
  __builtin_amdgcn_wmma_f32_16x16x32_f16(false,(A),false,(B),(short)0,(C),false,false)

static constexpr int      NPTS        = 4 * 262144;
static constexpr int      PTS_PER_IT  = 64;                // 4 M-tiles of 16
static constexpr int      ITERS       = NPTS / PTS_PER_IT; // 16384 exactly
static constexpr int      TBL         = 1 << 19;
static constexpr unsigned HMASK       = (unsigned)(TBL - 1);
static constexpr unsigned PRIME1      = 2654435761u;

// ---- LDS layout (bytes). Row strides padded (+8 halfs) for bank spread ----
static constexpr int S0 = 72;   // K-stride for layer0 operands (34 -> 64 pad, +8)
static constexpr int S1 = 264;  // K-stride for K=256 operands (+8)
static constexpr int WT0_OFF = 0;                          // 256 x 72 halfs
static constexpr int WT1_OFF = WT0_OFF + 256 * S0 * 2;     // 256 x 264 halfs
static constexpr int WT2_OFF = WT1_OFF + 256 * S1 * 2;     // 16  x 264 halfs
static constexpr int B0_OFF  = WT2_OFF + 16  * S1 * 2;     // 256 f32
static constexpr int B1_OFF  = B0_OFF  + 256 * 4;          // 256 f32
static constexpr int ENC_OFF = B1_OFF  + 256 * 4;          // 64 x 72 halfs (shared)
static constexpr int A1_OFF  = ENC_OFF + 64 * S0 * 2;      // 64 x 264 halfs (shared)
static constexpr int A2_OFF  = A1_OFF  + 64 * S1 * 2;      // 64 x 264 halfs (shared)
static constexpr int SMEM_TOTAL = A2_OFF + 64 * S1 * 2;    // = 259,328 B (< 320 KB)

// 16-half fragment slice: 8 contiguous halfs at p and 8 at p+16
// (ISA 16-bit A/B layout: lane holds K {0..7,16..23} or {8..15,24..31}).
__device__ __forceinline__ v16h frag_ld(const _Float16* p) {
  h8 lo = *(const h8*)(p);
  h8 hi = *(const h8*)(p + 16);
  return __builtin_shufflevector(lo, hi, 0,1,2,3,4,5,6,7,8,9,10,11,12,13,14,15);
}

__global__ __launch_bounds__(128, 1)
void ngp_fused_kernel(const float* __restrict__ coord,
                      const float* __restrict__ emb,
                      const float* __restrict__ W0, const float* __restrict__ b0,
                      const float* __restrict__ W1, const float* __restrict__ b1,
                      const float* __restrict__ W2, const float* __restrict__ b2,
                      float* __restrict__ out)
{
  extern __shared__ __align__(16) char smem[];
  _Float16* wt0 = (_Float16*)(smem + WT0_OFF);
  _Float16* wt1 = (_Float16*)(smem + WT1_OFF);
  _Float16* wt2 = (_Float16*)(smem + WT2_OFF);
  float*    b0s = (float*)(smem + B0_OFF);
  float*    b1s = (float*)(smem + B1_OFF);
  _Float16* encB = (_Float16*)(smem + ENC_OFF);   // 64 x S0, shared
  _Float16* a1   = (_Float16*)(smem + A1_OFF);    // 64 x S1, shared
  _Float16* a2   = (_Float16*)(smem + A2_OFF);    // 64 x S1, shared

  const int tid = threadIdx.x;

  // ---- one-time per block: stage weights transposed (N-major) as f16 ----
  for (int i = tid; i < 256 * 64; i += 128) {           // Wt0[n][k] = W0[k][n]
    int n = i >> 6, k = i & 63;
    wt0[n * S0 + k] = (k < 34) ? (_Float16)W0[k * 256 + n] : (_Float16)0.0f;
  }
  for (int i = tid; i < 256 * 256; i += 128) {          // Wt1[n][k] = W1[k][n]
    int n = i >> 8, k = i & 255;
    wt1[n * S1 + k] = (_Float16)W1[k * 256 + n];
  }
  for (int i = tid; i < 16 * 256; i += 128) {           // Wt2[n][k] = W2[k][n], N pad 3->16
    int n = i >> 8, k = i & 255;
    wt2[n * S1 + k] = (n < 3) ? (_Float16)W2[k * 3 + n] : (_Float16)0.0f;
  }
  for (int i = tid; i < 256; i += 128) { b0s[i] = b0[i]; b1s[i] = b1[i]; }
  __syncthreads();

  const int lane = tid & 31;
  const int wv   = tid >> 5;           // wave id 0..3: owns N-tiles {4wv..4wv+3}
  const int row  = lane & 15;
  const int col  = lane & 15;
  const int koff = (lane >> 4) << 3;   // 0 or 8: K sub-slice AND D-row offset

  for (int it = blockIdx.x; it < ITERS; it += gridDim.x) {
    // ================= hash-grid encode (2 threads per point) =============
    {
      const int p  = (tid >> 1) & 63;     // point within this iteration
      const int hf = tid & 1;             // half: levels 0-7 or 8-15
      const int gp = it * PTS_PER_IT + p;
      float x = coord[gp * 2 + 0], y = coord[gp * 2 + 1];
      x = fminf(fmaxf(x, -1.0f), 1.0f);
      y = fminf(fmaxf(y, -1.0f), 1.0f);
      _Float16* er = encB + p * S0;
      if (hf == 0) { er[0] = (_Float16)x; er[1] = (_Float16)y; }
      else {
        #pragma unroll
        for (int k = 34; k < 64; ++k) er[k] = (_Float16)0.0f;  // K pad
      }
      const int l0 = hf * 8;
      #pragma unroll
      for (int li = 0; li < 8; ++li) {
        const int l = l0 + li;
        // res_l = floor(16 * 2^(l/3))  (b = 32^(1/15) = 2^(1/3))
        const float res = floorf(16.0f * exp2f((float)l * (1.0f / 3.0f)));
        const float s = res * 0.5f;
        const float fx = (x + 1.0f) * s, fy = (y + 1.0f) * s;
        const float bx = floorf(fx),  by = floorf(fy);
        const float wx = fx - bx,     wy = fy - by;
        const unsigned ux = (unsigned)(int)bx, uy = (unsigned)(int)by;
        const float* tb = emb + (size_t)l * (size_t)TBL * 2;
        const unsigned h00 = (ux        ^ ( uy       * PRIME1)) & HMASK;
        const unsigned h01 = (ux        ^ ((uy + 1u) * PRIME1)) & HMASK;
        const unsigned h10 = ((ux + 1u) ^ ( uy       * PRIME1)) & HMASK;
        const unsigned h11 = ((ux + 1u) ^ ((uy + 1u) * PRIME1)) & HMASK;
        const float2 e00 = *(const float2*)(tb + (size_t)h00 * 2);
        const float2 e01 = *(const float2*)(tb + (size_t)h01 * 2);
        const float2 e10 = *(const float2*)(tb + (size_t)h10 * 2);
        const float2 e11 = *(const float2*)(tb + (size_t)h11 * 2);
        const float owx = 1.0f - wx, owy = 1.0f - wy;
        const float f0 = (e00.x * owx + e10.x * wx) * owy + (e01.x * owx + e11.x * wx) * wy;
        const float f1 = (e00.y * owx + e10.y * wx) * owy + (e01.y * owx + e11.y * wx) * wy;
        er[2 + 2 * l] = (_Float16)f0;
        er[3 + 2 * l] = (_Float16)f1;
      }
    }
    __syncthreads();

    // ===== layer 0: [64x64] @ [64x256]; wave owns 4 N-tiles, A resident ====
    {
      v16h Ae[4][2];
      #pragma unroll
      for (int m = 0; m < 4; ++m) {
        const _Float16* ap = encB + (m * 16 + row) * S0 + koff;
        Ae[m][0] = frag_ld(ap);
        Ae[m][1] = frag_ld(ap + 32);
      }
      const _Float16* bp = wt0 + ((4 * wv) * 16 + col) * S0 + koff;
      v16h B0 = frag_ld(bp), B1 = frag_ld(bp + 32);
      #pragma unroll
      for (int ni = 0; ni < 4; ++ni) {
        const int n = 4 * wv + ni;
        const v16h Bc0 = B0, Bc1 = B1;
        if (ni < 3) {                       // prefetch next n's B fragments
          const _Float16* bpn = wt0 + ((n + 1) * 16 + col) * S0 + koff;
          B0 = frag_ld(bpn); B1 = frag_ld(bpn + 32);
        }
        v8f acc[4];
        #pragma unroll
        for (int m = 0; m < 4; ++m) {
          acc[m] = (v8f){};
          acc[m] = WMMA_F16(Ae[m][0], Bc0, acc[m]);
          acc[m] = WMMA_F16(Ae[m][1], Bc1, acc[m]);
        }
        const float bias = b0s[n * 16 + col];
        #pragma unroll
        for (int m = 0; m < 4; ++m)
          #pragma unroll
          for (int r = 0; r < 8; ++r) {
            const float v = fmaxf(acc[m][r] + bias, 0.0f);
            a1[(m * 16 + r + koff) * S1 + n * 16 + col] = (_Float16)v;
          }
      }
    }
    __syncthreads();

    // ===== layer 1: [64x256] @ [256x256]; 2 resident M-tiles per pass =====
    #pragma unroll
    for (int mp = 0; mp < 2; ++mp) {
      const int m0 = 2 * mp, m1 = 2 * mp + 1;
      v16h A0[8], A1[8];
      {
        const _Float16* ap0 = a1 + (m0 * 16 + row) * S1 + koff;
        const _Float16* ap1 = a1 + (m1 * 16 + row) * S1 + koff;
        #pragma unroll
        for (int k = 0; k < 8; ++k) { A0[k] = frag_ld(ap0 + 32 * k);
                                      A1[k] = frag_ld(ap1 + 32 * k); }
      }
      #pragma unroll
      for (int ni = 0; ni < 4; ++ni) {
        const int n = 4 * wv + ni;
        const _Float16* bp = wt1 + (n * 16 + col) * S1 + koff;
        v8f acc0 = {}, acc1 = {};
        v16h B = frag_ld(bp);
        #pragma unroll
        for (int k = 0; k < 8; ++k) {
          const v16h Bc = B;
          if (k < 7) B = frag_ld(bp + 32 * (k + 1));   // prefetch next k
          acc0 = WMMA_F16(A0[k], Bc, acc0);
          acc1 = WMMA_F16(A1[k], Bc, acc1);
        }
        const float bias = b1s[n * 16 + col];
        #pragma unroll
        for (int r = 0; r < 8; ++r) {
          const float v0 = fmaxf(acc0[r] + bias, 0.0f);
          const float v1 = fmaxf(acc1[r] + bias, 0.0f);
          a2[(m0 * 16 + r + koff) * S1 + n * 16 + col] = (_Float16)v0;
          a2[(m1 * 16 + r + koff) * S1 + n * 16 + col] = (_Float16)v1;
        }
      }
    }
    __syncthreads();

    // ===== layer 2: [16x256] @ [256x16(pad of 3)]; wave owns M-tile wv ====
    {
      const _Float16* ap = a2 + (wv * 16 + row) * S1 + koff;
      const _Float16* bp = wt2 + col * S1 + koff;
      v8f acc = {};
      v16h B = frag_ld(bp);
      #pragma unroll
      for (int k = 0; k < 8; ++k) {
        const v16h Ak = frag_ld(ap + 32 * k);
        const v16h Bc = B;
        if (k < 7) B = frag_ld(bp + 32 * (k + 1));
        acc = WMMA_F16(Ak, Bc, acc);
      }
      if (col < 3) {
        const float bias = b2[col];
        const int gp0 = it * PTS_PER_IT + wv * 16 + koff;
        #pragma unroll
        for (int r = 0; r < 8; ++r) {
          const float v = acc[r] + bias;
          out[(size_t)(gp0 + r) * 3 + col] = 1.0f / (1.0f + expf(-v));
        }
      }
    }
    __syncthreads();   // protect enc/act staging before next iteration
  }
}

extern "C" void kernel_launch(void* const* d_in, const int* in_sizes, int n_in,
                              void* d_out, int out_size, void* d_ws, size_t ws_size,
                              hipStream_t stream) {
  (void)in_sizes; (void)n_in; (void)d_ws; (void)ws_size; (void)out_size;
  const float* coord = (const float*)d_in[0];
  const float* emb   = (const float*)d_in[1];
  const float* W0    = (const float*)d_in[2];
  const float* b0    = (const float*)d_in[3];
  const float* W1    = (const float*)d_in[4];
  const float* b1    = (const float*)d_in[5];
  const float* W2    = (const float*)d_in[6];
  const float* b2    = (const float*)d_in[7];
  float* out = (float*)d_out;

  (void)hipFuncSetAttribute((const void*)ngp_fused_kernel,
                            hipFuncAttributeMaxDynamicSharedMemorySize, SMEM_TOTAL);

  ngp_fused_kernel<<<dim3(512), dim3(128), SMEM_TOTAL, stream>>>(
      coord, emb, W0, b0, W1, b1, W2, b2, out);
}
